// RNN_2997887172658
// MI455X (gfx1250) — matrix operand order repaired and checked by reading, not compiled, not run on hardware
//
#include <hip/hip_runtime.h>
#include <math.h>

#define B_ROWS 4096
#define NS 1024
#define NC 256

typedef __attribute__((ext_vector_type(16))) __bf16 v16bf;
typedef __attribute__((ext_vector_type(8)))  __bf16 v8bf;
typedef __attribute__((ext_vector_type(4)))  __bf16 v4bf;
typedef __attribute__((ext_vector_type(8)))  float  v8f;

#define WMMA_BF16 __builtin_amdgcn_wmma_f32_16x16x32_bf16

__device__ __forceinline__ float apply_ep(float v, int ep) {
    if (ep == 1) return v > 0.0f ? v : 0.0f;   // relu
    if (ep == 2) return tanhf(v);              // tanh
    return v;
}

// ---- fragment loaders -------------------------------------------------------
// A 16x32 bf16 layout: lanes<16 hold K 0-7 (v0-3) and 16-23 (v4-7);
//                      lanes>=16 hold K 8-15 and 24-31.
__device__ __forceinline__ v16bf load_a_frag(const float* __restrict__ row,
                                             int k0, int half, float s) {
    v16bf a;
#pragma unroll
    for (int i = 0; i < 16; ++i) {
        const int ka = k0 + (i & 7) + half * 8 + ((i >= 8) ? 16 : 0);
        a[i] = (__bf16)(row[ka] * s);
    }
    return a;
}
__device__ __forceinline__ v16bf load_a_frag(const __bf16* __restrict__ row,
                                             int k0, int half, float) {
    // two aligned 16-byte loads (8 contiguous bf16 each)
    const v8bf lo = *(const v8bf*)(row + k0 + half * 8);
    const v8bf hi = *(const v8bf*)(row + k0 + 16 + half * 8);
    return __builtin_shufflevector(lo, hi, 0,1,2,3,4,5,6,7,8,9,10,11,12,13,14,15);
}
// B 32x16 bf16 layout: same as A: lanes<16 hold K 0-7 and 16-23, lanes>=16 hold K 8-15 and 24-31.
__device__ __forceinline__ v16bf load_b_frag(const float* __restrict__ row,
                                             int k0, int half) {
    v16bf b;
#pragma unroll
    for (int i = 0; i < 16; ++i) b[i] = (__bf16)row[k0 + (i & 7) + half * 8 + ((i >= 8) ? 16 : 0)];
    return b;
}
__device__ __forceinline__ v16bf load_b_frag(const __bf16* __restrict__ row,
                                             int k0, int half) {
    const v8bf lo = *(const v8bf*)(row + k0 + half * 8);
    const v8bf hi = *(const v8bf*)(row + k0 + 16 + half * 8);
    return __builtin_shufflevector(lo, hi, 0,1,2,3,4,5,6,7,8,9,10,11,12,13,14,15);
}

// C[M,N] = ep( concat_k(A0..A3)[M, nparts*Kpart] @ W[N, nparts*Kpart]^T + bias )
// T = __bf16 (pre-staged operands, pure b128 loads) or float (gather + cvt,
// optional per-(row,part) ascale). Wave tile (16*MT) x 64 -> MT*4 WMMAs/K-step.
// Block: 8 waves in 2x4 -> block tile (32*MT) x 256.
template <typename T, int MT>
__global__ __launch_bounds__(256) void gemm_wmma(
    const T* __restrict__ A0, const T* __restrict__ A1,
    const T* __restrict__ A2, const T* __restrict__ A3,
    int nparts, int Kpart,
    const T* __restrict__ W, const float* __restrict__ bias,
    const float* __restrict__ ascale,     // honored only for T=float
    float* __restrict__ Cf, __bf16* __restrict__ Cb,
    int M, int N, int ep)
{
    const int lane = threadIdx.x & 31;
    const int wave = threadIdx.x >> 5;
    const int half = lane >> 4;
    const int lm   = lane & 15;
    const int m0 = blockIdx.x * (32 * MT) + (wave >> 2) * (16 * MT);
    const int n0 = blockIdx.y * 256 + (wave & 3) * 64;
    const int ldw = nparts * Kpart;

    v8f acc[MT][4];
#pragma unroll
    for (int mt = 0; mt < MT; ++mt)
#pragma unroll
        for (int nt = 0; nt < 4; ++nt) acc[mt][nt] = (v8f){};

    for (int p = 0; p < nparts; ++p) {
        // SSA select chain (not an indexed array) keeps loads in global AS.
        const T* Ap = (p == 0) ? A0 : (p == 1) ? A1 : (p == 2) ? A2 : A3;
        const int kb = p * Kpart;
        const T* arow[MT];
        float s[MT];
#pragma unroll
        for (int mt = 0; mt < MT; ++mt) {
            arow[mt] = Ap + (size_t)(m0 + mt * 16 + lm) * Kpart;
            s[mt] = ascale ? ascale[(size_t)(m0 + mt * 16 + lm) * nparts + p] : 1.0f;
        }
        const T* brow[4];
#pragma unroll
        for (int nt = 0; nt < 4; ++nt)
            brow[nt] = W + (size_t)(n0 + nt * 16 + lm) * ldw + kb;

        for (int k0 = 0; k0 < Kpart; k0 += 32) {
            if (k0 + 32 < Kpart) {   // uniform; speculative prefetch of next K tile
#pragma unroll
                for (int mt = 0; mt < MT; ++mt) __builtin_prefetch(arow[mt] + k0 + 32, 0, 1);
#pragma unroll
                for (int nt = 0; nt < 4; ++nt)  __builtin_prefetch(brow[nt] + k0 + 32, 0, 1);
            }
            v16bf a[MT], b[4];
#pragma unroll
            for (int mt = 0; mt < MT; ++mt) a[mt] = load_a_frag(arow[mt], k0, half, s[mt]);
#pragma unroll
            for (int nt = 0; nt < 4; ++nt)  b[nt] = load_b_frag(brow[nt], k0, half);
#pragma unroll
            for (int mt = 0; mt < MT; ++mt)
#pragma unroll
                for (int nt = 0; nt < 4; ++nt)
                    acc[mt][nt] = WMMA_BF16(false, a[mt], false, b[nt],
                                            (short)0, acc[mt][nt], false, false);
        }
    }

    // C/D f32 16x16 layout: row = vgpr + 8*half, col = lane&15.
#pragma unroll
    for (int mt = 0; mt < MT; ++mt)
#pragma unroll
        for (int nt = 0; nt < 4; ++nt)
#pragma unroll
            for (int r = 0; r < 8; ++r) {
                const int row = m0 + mt * 16 + r + 8 * half;
                const int col = n0 + nt * 16 + lm;
                const float v = apply_ep(acc[mt][nt][r] + bias[col], ep);
                if (Cf) Cf[(size_t)row * N + col] = v;
                if (Cb) Cb[(size_t)row * N + col] = (__bf16)v;
            }
}

// ---- f32 -> bf16 bulk convert (4 elems/thread, vectorized) ----
__global__ __launch_bounds__(256) void cvt_bf16_kernel(
    const float* __restrict__ src, __bf16* __restrict__ dst, int n4)
{
    const int i = blockIdx.x * blockDim.x + threadIdx.x;
    if (i >= n4) return;
    const float4 v = ((const float4*)src)[i];
    v4bf o;
    o[0] = (__bf16)v.x; o[1] = (__bf16)v.y; o[2] = (__bf16)v.z; o[3] = (__bf16)v.w;
    ((v4bf*)dst)[i] = o;
}

// gated[b, part*NS+k] = bf16(rnn{part}t[b,k] * g[b,part])
__global__ __launch_bounds__(256) void gated_bf16_kernel(
    const float* __restrict__ r1, const float* __restrict__ r2,
    const float* __restrict__ r3, const float* __restrict__ r4,
    const float* __restrict__ g, __bf16* __restrict__ out)
{
    const size_t i = (size_t)blockIdx.x * blockDim.x + threadIdx.x;
    const size_t row = i / (4 * NS);
    const size_t j = i % (4 * NS);
    const int part = (int)(j / NS);
    const size_t k = j % NS;
    const float* src = (part == 0) ? r1 : (part == 1) ? r2 : (part == 2) ? r3 : r4;
    out[i] = (__bf16)(src[row * NS + k] * g[row * 4 + part]);
}

// ---- small N=2 projection + gumbel-hard one-hot + hierarchical blend ----
__global__ __launch_bounds__(256) void gate_kernel(
    const float* __restrict__ X, const float* __restrict__ W,
    const float* __restrict__ bias, const float* __restrict__ gn,
    const float* __restrict__ prev, const float* __restrict__ b1t,
    const float* __restrict__ b2t, float* __restrict__ out, int mode)
{
    const int b = blockIdx.x * blockDim.x + threadIdx.x;
    if (b >= B_ROWS) return;
    const float* x = X + (size_t)b * NS;
    float l0 = bias[0], l1 = bias[1];
    for (int k = 0; k < NS; ++k) {
        const float xv = x[k];
        l0 += xv * W[k];
        l1 += xv * W[NS + k];
    }
    l0 += gn[b * 2 + 0];
    l1 += gn[b * 2 + 1];
    const float h0 = (l0 >= l1) ? 1.0f : 0.0f;  // argmax ties -> index 0
    const float h1 = 1.0f - h0;
    float o0, o1;
    if (mode == 1) {
        o0 = h0; o1 = h1;
    } else if (mode == 2) {
        const float s1 = b1t[b * 2];
        o0 = s1 * h0 + (1.0f - s1) * prev[b * 2 + 0];
        o1 = s1 * h1 + (1.0f - s1) * prev[b * 2 + 1];
    } else {
        const float s1 = b1t[b * 2], s2 = b2t[b * 2];
        o0 = s1 * (s2 * h0 + (1.0f - s2) * prev[b * 2 + 0]) + (1.0f - s1) * prev[b * 2 + 0];
        o1 = s1 * (s2 * h1 + (1.0f - s2) * prev[b * 2 + 1]) + (1.0f - s1) * prev[b * 2 + 1];
    }
    out[b * 2 + 0] = o0;
    out[b * 2 + 1] = o1;
}

// g = sigmoid(hg @ W_g_hg^T + b), N=4
__global__ __launch_bounds__(256) void g_kernel(
    const float* __restrict__ hg, const float* __restrict__ W,
    const float* __restrict__ bias, float* __restrict__ g)
{
    const int b = blockIdx.x * blockDim.x + threadIdx.x;
    if (b >= B_ROWS) return;
    float acc0 = bias[0], acc1 = bias[1], acc2 = bias[2], acc3 = bias[3];
    const float* x = hg + (size_t)b * NS;
    for (int k = 0; k < NS; ++k) {
        const float xv = x[k];
        acc0 += xv * W[0 * NS + k];
        acc1 += xv * W[1 * NS + k];
        acc2 += xv * W[2 * NS + k];
        acc3 += xv * W[3 * NS + k];
    }
    g[b * 4 + 0] = 1.0f / (1.0f + expf(-acc0));
    g[b * 4 + 1] = 1.0f / (1.0f + expf(-acc1));
    g[b * 4 + 2] = 1.0f / (1.0f + expf(-acc2));
    g[b * 4 + 3] = 1.0f / (1.0f + expf(-acc3));
}

// Blend kernels: write f32 result (final output) and optional bf16 copy (GEMM input).
__global__ __launch_bounds__(256) void blend1_kernel(
    float* __restrict__ out, __bf16* __restrict__ outb,
    const float* __restrict__ x1, const float* __restrict__ x0,
    const float* __restrict__ b1)
{
    const size_t i = (size_t)blockIdx.x * blockDim.x + threadIdx.x;
    const float s = b1[(i / NS) * 2];
    const float v = s * x1[i] + (1.0f - s) * x0[i];
    out[i] = v;
    if (outb) outb[i] = (__bf16)v;
}

__global__ __launch_bounds__(256) void blend2_kernel(
    float* __restrict__ out, __bf16* __restrict__ outb,
    const float* __restrict__ rnn2, const float* __restrict__ b0,
    const float* __restrict__ b1v, const float* __restrict__ b1t,
    const float* __restrict__ b2)
{
    const size_t i = (size_t)blockIdx.x * blockDim.x + threadIdx.x;
    const size_t r = i / NS;
    const float s1 = b1t[r * 2], s2 = b2[r * 2];
    const float inner = (1.0f - s2) * b0[i] + s2 * b1v[i];
    const float v = (1.0f - s1) * rnn2[i] + s1 * inner;
    out[i] = v;
    if (outb) outb[i] = (__bf16)v;
}

__global__ __launch_bounds__(256) void blend3_kernel(
    float* __restrict__ out, __bf16* __restrict__ outb,
    const float* __restrict__ rnn3, const float* __restrict__ b1v,
    const float* __restrict__ b0, const float* __restrict__ b1t,
    const float* __restrict__ b2t, const float* __restrict__ b3)
{
    const size_t i = (size_t)blockIdx.x * blockDim.x + threadIdx.x;
    const size_t r = i / NS;
    const float s1 = b1t[r * 2], s2 = b2t[r * 2], s3 = b3[r * 2];
    const float inner = s3 * b1v[i] + (1.0f - s3) * b0[i];
    const float vv = rnn3[i];
    const float v = (1.0f - s1) * vv + s1 * ((1.0f - s2) * vv + s2 * inner);
    out[i] = v;
    if (outb) outb[i] = (__bf16)v;
}

__global__ __launch_bounds__(256) void blend4_kernel(
    float* __restrict__ out, __bf16* __restrict__ outb,
    const float* __restrict__ rnn4, const float* __restrict__ inner,
    const float* __restrict__ b1t, const float* __restrict__ b2t,
    const float* __restrict__ b3t)
{
    const size_t i = (size_t)blockIdx.x * blockDim.x + threadIdx.x;
    const size_t r = i / NS;
    const float s1 = b1t[r * 2], s2 = b2t[r * 2], s3 = b3t[r * 2];
    const float vv = rnn4[i];
    const float v = (1.0f - s1) * vv +
                    s1 * ((1.0f - s2) * vv + s2 * ((1.0f - s3) * vv + s3 * inner[i]));
    out[i] = v;
    if (outb) outb[i] = (__bf16)v;
}

// log_softmax over 256 columns; one 256-thread block per row.
__global__ __launch_bounds__(256) void logsoftmax_kernel(
    const float* __restrict__ logits, float* __restrict__ out)
{
    __shared__ float red[256];
    const int row = blockIdx.x;
    const int t = threadIdx.x;
    const float v = logits[(size_t)row * NC + t];
    red[t] = v;
    __syncthreads();
    for (int s = 128; s > 0; s >>= 1) {
        if (t < s) red[t] = fmaxf(red[t], red[t + s]);
        __syncthreads();
    }
    const float mx = red[0];
    __syncthreads();
    red[t] = expf(v - mx);
    __syncthreads();
    for (int s = 128; s > 0; s >>= 1) {
        if (t < s) red[t] += red[t + s];
        __syncthreads();
    }
    out[(size_t)row * NC + t] = v - (mx + logf(red[0]));
}

extern "C" void kernel_launch(void* const* d_in, const int* in_sizes, int n_in,
                              void* d_out, int out_size, void* d_ws, size_t ws_size,
                              hipStream_t stream) {
    // ---- inputs (setup_inputs order) ----
    const float* ct   = (const float*)d_in[0];
    const float* b1   = (const float*)d_in[1];
    const float* b2   = (const float*)d_in[2];
    const float* b3   = (const float*)d_in[3];
    const float* rnn1 = (const float*)d_in[4];
    const float* rnn2 = (const float*)d_in[5];
    const float* rnn3 = (const float*)d_in[6];
    const float* rnn4 = (const float*)d_in[7];
    const float* gn1  = (const float*)d_in[8];
    const float* gn2  = (const float*)d_in[9];
    const float* gn3  = (const float*)d_in[10];
    // d_in[11] = theta: tau>0 only scales logits before argmax -> no forward effect.
    const float* W_e_ct    = (const float*)d_in[12]; const float* b_e_ct    = (const float*)d_in[13];
    const float* W_rnn1_b0 = (const float*)d_in[14]; const float* b_rnn1_b0 = (const float*)d_in[15];
    const float* W_rnn1_b1 = (const float*)d_in[16]; const float* b_rnn1_b1 = (const float*)d_in[17];
    const float* W_b1      = (const float*)d_in[18]; const float* b_b1      = (const float*)d_in[19];
    const float* W_rnn2_b0 = (const float*)d_in[20]; const float* b_rnn2_b0 = (const float*)d_in[21];
    const float* W_rnn2_b1 = (const float*)d_in[22]; const float* b_rnn2_b1 = (const float*)d_in[23];
    const float* W_b2      = (const float*)d_in[24]; const float* b_b2      = (const float*)d_in[25];
    const float* W_rnn3_b0 = (const float*)d_in[26]; const float* b_rnn3_b0 = (const float*)d_in[27];
    const float* W_rnn3_b1 = (const float*)d_in[28]; const float* b_rnn3_b1 = (const float*)d_in[29];
    const float* W_b3      = (const float*)d_in[30]; const float* b_b3      = (const float*)d_in[31];
    const float* W_rnn4    = (const float*)d_in[32]; const float* b_rnn4    = (const float*)d_in[33];
    const float* W_hg      = (const float*)d_in[34]; const float* b_hg      = (const float*)d_in[35];
    const float* W_g       = (const float*)d_in[36]; const float* b_g       = (const float*)d_in[37];
    const float* W_hctpi   = (const float*)d_in[38]; const float* b_hctpi   = (const float*)d_in[39];
    const float* W_ctpi    = (const float*)d_in[40]; const float* b_ctpi    = (const float*)d_in[41];
    (void)in_sizes; (void)n_in; (void)out_size;

    // ---- outputs: (ctpi, b1t, b2t, b3t, rnn1t, rnn2t, rnn3t, rnn4t) flat ----
    float* out   = (float*)d_out;
    float* ctpi  = out;
    float* b1t   = ctpi + (size_t)B_ROWS * NC;
    float* b2t   = b1t + (size_t)B_ROWS * 2;
    float* b3t   = b2t + (size_t)B_ROWS * 2;
    float* rnn1t = b3t + (size_t)B_ROWS * 2;
    float* rnn2t = rnn1t + (size_t)B_ROWS * NS;
    float* rnn3t = rnn2t + (size_t)B_ROWS * NS;
    float* rnn4t = rnn3t + (size_t)B_ROWS * NS;

    const size_t BN = (size_t)B_ROWS * NS;
    const size_t BC = (size_t)B_ROWS * NC;

    const dim3 blk(256);
    const int ew_blocks = (int)((BN) / 256);
    const int row_blocks = B_ROWS / 256;

    // workspace requirement of the bf16-staged fast path
    const size_t f32_elems = 2 * BN + BC + (size_t)B_ROWS * 4;
    const size_t wbf_elems = (size_t)NS * NC + 7 * (size_t)NS * 2 * NS +
                             2 * (size_t)NS * 4 * NS + (size_t)NC * NS;
    const size_t abf_elems = BC + 4 * BN /*rnn in / gated*/ + BN /*ct_e*/ +
                             4 * BN /*rnnXt*/ + BN /*hctpi*/;
    const size_t need_full = f32_elems * 4 + (wbf_elems + abf_elems) * 2;

    if (ws_size >= need_full) {
        // ================= bf16-staged fast path =================
        float* ws1    = (float*)d_ws;              // B x NS f32 scratch
        float* ws2    = ws1 + BN;                  // B x NS f32 scratch
        float* logits = ws2 + BN;                  // B x NC
        float* gbuf   = logits + BC;               // B x 4
        __bf16* bp = (__bf16*)(gbuf + (size_t)B_ROWS * 4);
        __bf16* ct_bf   = bp; bp += BC;
        __bf16* r1_bf   = bp; bp += BN;
        __bf16* r2_bf   = bp; bp += BN;
        __bf16* r3_bf   = bp; bp += BN;
        __bf16* r4_bf   = bp; bp += BN;
        __bf16* gated_bf = r1_bf;                  // reuse rnn-input block (4*BN)
        __bf16* cte_bf  = bp; bp += BN;
        __bf16* t1_bf   = bp; bp += BN;
        __bf16* t2_bf   = bp; bp += BN;
        __bf16* t3_bf   = bp; bp += BN;
        __bf16* t4_bf   = bp; bp += BN;
        __bf16* hct_bf  = bp; bp += BN;
        __bf16* Wb_e    = bp; bp += (size_t)NS * NC;
        __bf16* Wb_r1b0 = bp; bp += (size_t)NS * 2 * NS;
        __bf16* Wb_r1b1 = bp; bp += (size_t)NS * 2 * NS;
        __bf16* Wb_r2b0 = bp; bp += (size_t)NS * 2 * NS;
        __bf16* Wb_r2b1 = bp; bp += (size_t)NS * 2 * NS;
        __bf16* Wb_r3b0 = bp; bp += (size_t)NS * 2 * NS;
        __bf16* Wb_r3b1 = bp; bp += (size_t)NS * 2 * NS;
        __bf16* Wb_r4   = bp; bp += (size_t)NS * 2 * NS;
        __bf16* Wb_hg   = bp; bp += (size_t)NS * 4 * NS;
        __bf16* Wb_hct  = bp; bp += (size_t)NS * 4 * NS;
        __bf16* Wb_ctpi = bp; bp += (size_t)NC * NS;

        auto cvt = [&](const float* s, __bf16* d, size_t n) {
            cvt_bf16_kernel<<<(int)(n / 4 / 256), blk, 0, stream>>>(s, d, (int)(n / 4));
        };
        // stage weights + input activations as bf16 (pure-bf16 GEMM operands)
        cvt(W_e_ct, Wb_e, (size_t)NS * NC);
        cvt(W_rnn1_b0, Wb_r1b0, (size_t)NS * 2 * NS);
        cvt(W_rnn1_b1, Wb_r1b1, (size_t)NS * 2 * NS);
        cvt(W_rnn2_b0, Wb_r2b0, (size_t)NS * 2 * NS);
        cvt(W_rnn2_b1, Wb_r2b1, (size_t)NS * 2 * NS);
        cvt(W_rnn3_b0, Wb_r3b0, (size_t)NS * 2 * NS);
        cvt(W_rnn3_b1, Wb_r3b1, (size_t)NS * 2 * NS);
        cvt(W_rnn4, Wb_r4, (size_t)NS * 2 * NS);
        cvt(W_hg, Wb_hg, (size_t)NS * 4 * NS);
        cvt(W_hctpi, Wb_hct, (size_t)NS * 4 * NS);
        cvt(W_ctpi, Wb_ctpi, (size_t)NC * NS);
        cvt(ct, ct_bf, BC);
        cvt(rnn1, r1_bf, BN);
        cvt(rnn2, r2_bf, BN);
        cvt(rnn3, r3_bf, BN);
        cvt(rnn4, r4_bf, BN);

        const dim3 gN(B_ROWS / 128, 1024 / 256);   // bf16 tile: 128x256 block
        const dim3 gC(B_ROWS / 128, 256 / 256);
        const __bf16* nb = nullptr;

        // 1) ct_e(bf16) = relu(ct @ W_e_ct^T + b)
        gemm_wmma<__bf16, 4><<<gN, blk, 0, stream>>>(ct_bf, nb, nb, nb, 1, NC,
            Wb_e, b_e_ct, nullptr, nullptr, cte_bf, B_ROWS, NS, 1);
        // 2/3) rnn1 branches (f32 for blending)
        gemm_wmma<__bf16, 4><<<gN, blk, 0, stream>>>(r2_bf, cte_bf, nb, nb, 2, NS,
            Wb_r1b1, b_rnn1_b1, nullptr, ws1, nullptr, B_ROWS, NS, 2);
        gemm_wmma<__bf16, 4><<<gN, blk, 0, stream>>>(r1_bf, cte_bf, nb, nb, 2, NS,
            Wb_r1b0, b_rnn1_b0, nullptr, ws2, nullptr, B_ROWS, NS, 2);
        // 4) rnn1t (f32 + bf16)
        blend1_kernel<<<ew_blocks, blk, 0, stream>>>(rnn1t, t1_bf, ws1, ws2, b1);
        // 5) b1t
        gate_kernel<<<row_blocks, blk, 0, stream>>>(rnn1t, W_b1, b_b1, gn1,
                                                    nullptr, nullptr, nullptr, b1t, 1);
        // 6/7) rnn2 branches
        gemm_wmma<__bf16, 4><<<gN, blk, 0, stream>>>(r3_bf, t1_bf, nb, nb, 2, NS,
            Wb_r2b1, b_rnn2_b1, nullptr, ws1, nullptr, B_ROWS, NS, 2);
        gemm_wmma<__bf16, 4><<<gN, blk, 0, stream>>>(r2_bf, t1_bf, nb, nb, 2, NS,
            Wb_r2b0, b_rnn2_b0, nullptr, ws2, nullptr, B_ROWS, NS, 2);
        // 8) rnn2t
        blend2_kernel<<<ew_blocks, blk, 0, stream>>>(rnn2t, t2_bf, rnn2, ws2, ws1, b1t, b2);
        // 9) b2t
        gate_kernel<<<row_blocks, blk, 0, stream>>>(rnn2t, W_b2, b_b2, gn2,
                                                    b2, b1t, nullptr, b2t, 2);
        // 10/11) rnn3 branches
        gemm_wmma<__bf16, 4><<<gN, blk, 0, stream>>>(r4_bf, t2_bf, nb, nb, 2, NS,
            Wb_r3b1, b_rnn3_b1, nullptr, ws1, nullptr, B_ROWS, NS, 2);
        gemm_wmma<__bf16, 4><<<gN, blk, 0, stream>>>(r3_bf, t2_bf, nb, nb, 2, NS,
            Wb_r3b0, b_rnn3_b0, nullptr, ws2, nullptr, B_ROWS, NS, 2);
        // 12) rnn3t
        blend3_kernel<<<ew_blocks, blk, 0, stream>>>(rnn3t, t3_bf, rnn3, ws1, ws2, b1t, b2t, b3);
        // 13) b3t
        gate_kernel<<<row_blocks, blk, 0, stream>>>(rnn3t, W_b3, b_b3, gn3,
                                                    b3, b1t, b2t, b3t, 3);
        // 14) rnn4_inner
        gemm_wmma<__bf16, 4><<<gN, blk, 0, stream>>>(r4_bf, t3_bf, nb, nb, 2, NS,
            Wb_r4, b_rnn4, nullptr, ws1, nullptr, B_ROWS, NS, 2);
        // 15) rnn4t
        blend4_kernel<<<ew_blocks, blk, 0, stream>>>(rnn4t, t4_bf, rnn4, ws1, b1t, b2t, b3t);
        // 16) hg(f32) = relu([t1..t4] @ W_hg^T + b)  (only feeds N=4 projection)
        gemm_wmma<__bf16, 4><<<gN, blk, 0, stream>>>(t1_bf, t2_bf, t3_bf, t4_bf, 4, NS,
            Wb_hg, b_hg, nullptr, ws1, nullptr, B_ROWS, NS, 1);
        // 17) g
        g_kernel<<<row_blocks, blk, 0, stream>>>(ws1, W_g, b_g, gbuf);
        // 17.5) gated bf16 (reuses rnn-input bf16 block; rnn inputs no longer needed)
        gated_bf16_kernel<<<(int)(4 * BN / 256), blk, 0, stream>>>(rnn1t, rnn2t, rnn3t,
                                                                   rnn4t, gbuf, gated_bf);
        // 18) hctpi(bf16) = relu(gated @ W_hctpi^T + b)
        gemm_wmma<__bf16, 4><<<gN, blk, 0, stream>>>(gated_bf, nb, nb, nb, 1, 4 * NS,
            Wb_hct, b_hctpi, nullptr, nullptr, hct_bf, B_ROWS, NS, 1);
        // 19) logits = hctpi @ W_ctpi^T + b
        gemm_wmma<__bf16, 4><<<gC, blk, 0, stream>>>(hct_bf, nb, nb, nb, 1, NS,
            Wb_ctpi, b_ctpi, nullptr, logits, nullptr, B_ROWS, NC, 0);
        // 20) ctpi
        logsoftmax_kernel<<<B_ROWS, blk, 0, stream>>>(logits, ctpi);
    } else {
        // ================= f32 on-the-fly fallback =================
        float* ws0    = (float*)d_ws;               // B x NS (ct_e, hg)
        float* ws1    = ws0 + BN;                   // B x NS
        float* ws2    = ws1 + BN;                   // B x NS
        float* logits = ws2 + BN;                   // B x NC
        float* gbuf   = logits + BC;                // B x 4
        const float* nf = nullptr;
        const dim3 gN(B_ROWS / 64, 1024 / 256);
        const dim3 gC(B_ROWS / 64, 256 / 256);

        gemm_wmma<float, 2><<<gN, blk, 0, stream>>>(ct, nf, nf, nf, 1, NC,
            W_e_ct, b_e_ct, nullptr, ws0, nullptr, B_ROWS, NS, 1);
        gemm_wmma<float, 2><<<gN, blk, 0, stream>>>(rnn2, ws0, nf, nf, 2, NS,
            W_rnn1_b1, b_rnn1_b1, nullptr, ws1, nullptr, B_ROWS, NS, 2);
        gemm_wmma<float, 2><<<gN, blk, 0, stream>>>(rnn1, ws0, nf, nf, 2, NS,
            W_rnn1_b0, b_rnn1_b0, nullptr, ws2, nullptr, B_ROWS, NS, 2);
        blend1_kernel<<<ew_blocks, blk, 0, stream>>>(rnn1t, nullptr, ws1, ws2, b1);
        gate_kernel<<<row_blocks, blk, 0, stream>>>(rnn1t, W_b1, b_b1, gn1,
                                                    nullptr, nullptr, nullptr, b1t, 1);
        gemm_wmma<float, 2><<<gN, blk, 0, stream>>>(rnn3, rnn1t, nf, nf, 2, NS,
            W_rnn2_b1, b_rnn2_b1, nullptr, ws1, nullptr, B_ROWS, NS, 2);
        gemm_wmma<float, 2><<<gN, blk, 0, stream>>>(rnn2, rnn1t, nf, nf, 2, NS,
            W_rnn2_b0, b_rnn2_b0, nullptr, ws2, nullptr, B_ROWS, NS, 2);
        blend2_kernel<<<ew_blocks, blk, 0, stream>>>(rnn2t, nullptr, rnn2, ws2, ws1, b1t, b2);
        gate_kernel<<<row_blocks, blk, 0, stream>>>(rnn2t, W_b2, b_b2, gn2,
                                                    b2, b1t, nullptr, b2t, 2);
        gemm_wmma<float, 2><<<gN, blk, 0, stream>>>(rnn4, rnn2t, nf, nf, 2, NS,
            W_rnn3_b1, b_rnn3_b1, nullptr, ws1, nullptr, B_ROWS, NS, 2);
        gemm_wmma<float, 2><<<gN, blk, 0, stream>>>(rnn3, rnn2t, nf, nf, 2, NS,
            W_rnn3_b0, b_rnn3_b0, nullptr, ws2, nullptr, B_ROWS, NS, 2);
        blend3_kernel<<<ew_blocks, blk, 0, stream>>>(rnn3t, nullptr, rnn3, ws1, ws2, b1t, b2t, b3);
        gate_kernel<<<row_blocks, blk, 0, stream>>>(rnn3t, W_b3, b_b3, gn3,
                                                    b3, b1t, b2t, b3t, 3);
        gemm_wmma<float, 2><<<gN, blk, 0, stream>>>(rnn4, rnn3t, nf, nf, 2, NS,
            W_rnn4, b_rnn4, nullptr, ws1, nullptr, B_ROWS, NS, 2);
        blend4_kernel<<<ew_blocks, blk, 0, stream>>>(rnn4t, nullptr, rnn4, ws1, b1t, b2t, b3t);
        gemm_wmma<float, 2><<<gN, blk, 0, stream>>>(rnn1t, rnn2t, rnn3t, rnn4t, 4, NS,
            W_hg, b_hg, nullptr, ws0, nullptr, B_ROWS, NS, 1);
        g_kernel<<<row_blocks, blk, 0, stream>>>(ws0, W_g, b_g, gbuf);
        gemm_wmma<float, 2><<<gN, blk, 0, stream>>>(rnn1t, rnn2t, rnn3t, rnn4t, 4, NS,
            W_hctpi, b_hctpi, gbuf, ws1, nullptr, B_ROWS, NS, 1);
        gemm_wmma<float, 2><<<gC, blk, 0, stream>>>(ws1, nf, nf, nf, 1, NS,
            W_ctpi, b_ctpi, nullptr, logits, nullptr, B_ROWS, NC, 0);
        logsoftmax_kernel<<<B_ROWS, blk, 0, stream>>>(logits, ctpi);
    }
}
